// xLSTMPositionNet_38328288149795
// MI455X (gfx1250) — matrix operand order repaired and checked by reading, not compiled, not run on hardware
//
#include <hip/hip_runtime.h>
#include <hip/hip_bf16.h>

// ---------------- problem constants ----------------
#define BB    128
#define TT    512
#define FF    256
#define HH    512
#define MT_P  4096          // (T*B)/16  m-tiles for projection
#define NT_P  32            // H/16      n-tiles
#define HPAD  520           // LDS row stride (elements) for staged h
#define NWG   128           // scan workgroups: 8 batch tiles x 16 col groups

typedef __attribute__((ext_vector_type(16))) __bf16 v16bf;
typedef __attribute__((ext_vector_type(8)))  __bf16 v8bf;
typedef __attribute__((ext_vector_type(8)))  float  v8f;

// ---------------- helpers ----------------
__device__ __forceinline__ unsigned short f2bf(float x) {
    unsigned u = __builtin_bit_cast(unsigned, x);
    unsigned r = u + 0x7fffu + ((u >> 16) & 1u);   // round-to-nearest-even
    return (unsigned short)(r >> 16);
}
__device__ __forceinline__ float bf2f(unsigned short s) {
    return __builtin_bit_cast(float, ((unsigned)s) << 16);
}
__device__ __forceinline__ v16bf make_frag(uint4 lo, uint4 hi) {
    v8bf l = __builtin_bit_cast(v8bf, lo);
    v8bf h = __builtin_bit_cast(v8bf, hi);
    return __builtin_shufflevector(l, h, 0,1,2,3,4,5,6,7,8,9,10,11,12,13,14,15);
}
__device__ __forceinline__ v8f wmma_bf16(v16bf a, v16bf b, v8f c) {
    return __builtin_amdgcn_wmma_f32_16x16x32_bf16(
        false, a, false, b, (short)0, c, false, false);
}
// async global -> LDS copy of 16 bytes per lane (ASYNCcnt tracked)
__device__ __forceinline__ void async_g2l_b128(unsigned ldsOff,
                                               const void* gaddr) {
    asm volatile("global_load_async_to_lds_b128 %0, %1, off"
                 :: "v"(ldsOff), "v"((unsigned long long)(uintptr_t)gaddr)
                 : "memory");
}
__device__ __forceinline__ void wait_asynccnt0() {
    asm volatile("s_wait_asynccnt 0x0" ::: "memory");
}

// ---------------- convert kernels ----------------
// x [B,T,F] f32 -> xT [T,B,F] bf16
__global__ __launch_bounds__(256) void k_cvt_x(const float* __restrict__ x,
                                               unsigned short* __restrict__ xT) {
    int idx = blockIdx.x * 256 + threadIdx.x;          // < B*T*F
    int f  = idx % FF;
    int bt = idx / FF;
    int t  = bt % TT;
    int b  = bt / TT;
    xT[(t * BB + b) * FF + f] = f2bf(x[idx]);
}

// src [rows][cols] f32 -> dst [cols][rows] bf16   (transpose + convert)
__global__ __launch_bounds__(256) void k_cvt_tr(const float* __restrict__ src,
                                                unsigned short* __restrict__ dst,
                                                int rows, int cols) {
    int idx = blockIdx.x * 256 + threadIdx.x;
    if (idx >= rows * cols) return;
    int r = idx / cols, c = idx % cols;
    dst[c * rows + r] = f2bf(src[idx]);
}

// zero the per-step barrier counters (must run every launch)
__global__ __launch_bounds__(256) void k_zero(unsigned* __restrict__ p, int n) {
    int idx = blockIdx.x * 256 + threadIdx.x;
    if (idx < n) p[idx] = 0u;
}

// ---------------- phase 1: input projections ----------------
// Xproj[gate][mtile][ntile][256] bf16 fragments; rows indexed (t*B + b)
__global__ __launch_bounds__(256) void k_proj(
        const unsigned short* __restrict__ xT,   // [T*B][F] bf16
        const unsigned short* __restrict__ WT,   // [4][H][F] bf16 (transposed)
        const float* __restrict__ bf_, const float* __restrict__ bi_,
        const float* __restrict__ bo_, const float* __restrict__ bz_,
        unsigned short* __restrict__ Xproj) {
    const int gate = blockIdx.z;
    const int lane = threadIdx.x & 31;
    const int wave = threadIdx.x >> 5;
    const int lm   = lane & 15;
    const int lh   = lane >> 4;
    const int mt     = blockIdx.y * 2 + (wave >> 2);
    const int ntBase = blockIdx.x * 16 + (wave & 3) * 4;
    const float* bias = (gate == 0) ? bf_ : (gate == 1) ? bi_ : (gate == 2) ? bo_ : bz_;

    v8f acc[4] = {};
#pragma unroll
    for (int kk = 0; kk < FF / 32; ++kk) {
        const int kb = kk * 32 + lh * 16;
        const unsigned short* arow = xT + (size_t)(mt * 16 + lm) * FF + kb;
        v16bf a = make_frag(*(const uint4*)arow, *(const uint4*)(arow + 8));
#pragma unroll
        for (int c4 = 0; c4 < 4; ++c4) {
            const int n = (ntBase + c4) * 16 + lm;
            const unsigned short* brow =
                WT + ((size_t)gate * HH + n) * FF + kk * 32 + lh * 16;
            v16bf b = make_frag(*(const uint4*)brow, *(const uint4*)(brow + 8));
            acc[c4] = wmma_bf16(a, b, acc[c4]);
        }
    }
#pragma unroll
    for (int c4 = 0; c4 < 4; ++c4) {
        const int nt = ntBase + c4;
        const float bv = bias[nt * 16 + lm];
        unsigned short u[8];
#pragma unroll
        for (int e = 0; e < 8; ++e) u[e] = f2bf(acc[c4][e] + bv);
        uint4 st;
        st.x = (unsigned)u[0] | ((unsigned)u[1] << 16);
        st.y = (unsigned)u[2] | ((unsigned)u[3] << 16);
        st.z = (unsigned)u[4] | ((unsigned)u[5] << 16);
        st.w = (unsigned)u[6] | ((unsigned)u[7] << 16);
        size_t tile = ((size_t)gate * MT_P + mt) * NT_P + nt;
        *(uint4*)(Xproj + tile * 256 + lane * 8) = st;
    }
}

// ---------------- phase 2: persistent recurrent scan ----------------
// 128 WGs x 64 threads: WG = (batch tile mt, column group cg of 32 cols).
// R slice (4 gates x 32 cols x 512 K bf16 = 128 KB) lives in LDS for the
// whole scan; per step only the tiny h state moves through L2.
__global__ __launch_bounds__(64) void k_scan(
        const unsigned short* __restrict__ RT,     // [4][H][H] bf16 (transposed)
        const unsigned short* __restrict__ Xproj,  // fragment-ordered tiles
        const float* __restrict__ fc_w, const float* __restrict__ fc_b,
        unsigned short* __restrict__ h0,           // [8][16][512] bf16
        unsigned short* __restrict__ h1,           // [8][16][512] bf16
        unsigned* __restrict__ cnt,                // [T] step barrier counters
        float* __restrict__ out) {
    extern __shared__ unsigned short smem[];
    unsigned short* rbuf   = smem;            // 2 tiles*4 gates*16 kk*512 = 65536
    unsigned short* hstage = smem + 65536;    // 16*HPAD = 8320

    const int mt   = blockIdx.x >> 4;         // batch tile 0..7
    const int cg   = blockIdx.x & 15;         // column group 0..15
    const int lane = threadIdx.x & 31;
    const int wav  = threadIdx.x >> 5;        // 0..1
    const int lm   = lane & 15;
    const int lh   = lane >> 4;
    const int jt   = cg * 2 + wav;            // global col tile 0..31

    // ---- one-time preload of this WG's R slice into LDS (fragment order) ----
#pragma unroll
    for (int gg = 0; gg < 4; ++gg) {
        for (int kk = 0; kk < HH / 32; ++kk) {
            const unsigned short* src =
                RT + ((size_t)gg * HH + jt * 16 + lm) * HH + kk * 32 + lh * 16;
            unsigned short* dst = rbuf + ((size_t)((wav * 4 + gg) * 16 + kk)) * 512 + lane * 16;
            *(uint4*)dst       = *(const uint4*)src;
            *(uint4*)(dst + 8) = *(const uint4*)(src + 8);
        }
    }
    __syncthreads();

    v8f cs = {}, ns = {};                     // persistent c/n for this wave's tile

    for (int t = 0; t < TT; ++t) {
        const unsigned short* hin = (t & 1) ? h1 : h0;
        unsigned short*      hout = (t & 1) ? h0 : h1;
        const int mtp = t * (BB / 16) + mt;   // projection m-tile

        // stage h[mt] (16x512 bf16) into LDS via async global->LDS copies
        if (t > 0) {
            for (int i = threadIdx.x; i < 16 * (HH / 8); i += 64) {
                int r = i >> 6, c8 = (i & 63) * 8;   // 8 elems (16 B) per copy
                unsigned dstOff =
                    (unsigned)(uintptr_t)(hstage + r * HPAD + c8);
                async_g2l_b128(dstOff, hin + ((size_t)mt * 16 + r) * HH + c8);
            }
            wait_asynccnt0();
        }
        __syncthreads();

        // accumulators init = precomputed input projection fragments
        v8f acc[4];
#pragma unroll
        for (int gg = 0; gg < 4; ++gg) {
            size_t tile = ((size_t)gg * MT_P + mtp) * NT_P + jt;
            uint4 q = *(const uint4*)(Xproj + tile * 256 + lane * 8);
            if (t + 1 < TT) {   // prefetch next step's fragment
                size_t tn = ((size_t)gg * MT_P + (mtp + BB / 16)) * NT_P + jt;
                __builtin_prefetch(Xproj + tn * 256 + lane * 8, 0, 1);
            }
            v8f a;
            a[0] = bf2f((unsigned short)(q.x & 0xffff));
            a[1] = bf2f((unsigned short)(q.x >> 16));
            a[2] = bf2f((unsigned short)(q.y & 0xffff));
            a[3] = bf2f((unsigned short)(q.y >> 16));
            a[4] = bf2f((unsigned short)(q.z & 0xffff));
            a[5] = bf2f((unsigned short)(q.z >> 16));
            a[6] = bf2f((unsigned short)(q.w & 0xffff));
            a[7] = bf2f((unsigned short)(q.w >> 16));
            acc[gg] = a;
        }

        if (t > 0) {
            for (int kk = 0; kk < HH / 32; ++kk) {
                const unsigned short* ar = hstage + lm * HPAD + kk * 32 + lh * 16;
                v16bf a = make_frag(*(const uint4*)ar, *(const uint4*)(ar + 8));
#pragma unroll
                for (int gg = 0; gg < 4; ++gg) {
                    const unsigned short* br =
                        rbuf + ((size_t)((wav * 4 + gg) * 16 + kk)) * 512 + lane * 16;
                    v16bf b = make_frag(*(const uint4*)br, *(const uint4*)(br + 8));
                    acc[gg] = wmma_bf16(a, b, acc[gg]);
                }
            }
        }

        // elementwise sLSTM cell update; write h slice to global double buffer
#pragma unroll
        for (int e = 0; e < 8; ++e) {
            float ft = acc[0][e];
            float it = acc[1][e];
            float o  = 1.0f / (1.0f + __expf(-acc[2][e]));
            float e2 = __expf(2.0f * acc[3][e]);
            float z  = (e2 - 1.0f) / (e2 + 1.0f);
            float fh = __expf(fminf(ft, 10.0f));
            float ih = __expf(fminf(it, 10.0f));
            float dn = fh + ih + 1e-8f;
            float f  = fh / dn;
            float i  = ih / dn;
            cs[e] = f * cs[e] + i * z;
            ns[e] = f * ns[e] + i;
            float h = o * (cs[e] / (ns[e] + 1e-8f));
            int row = e + 8 * lh;
            hout[((size_t)mt * 16 + row) * HH + jt * 16 + lm] = f2bf(h);
        }

        // ---- device-wide step barrier (all 128 WGs co-resident) ----
        __threadfence();                       // release h writes
        __syncthreads();
        if (threadIdx.x == 0) {
            __hip_atomic_fetch_add(&cnt[t], 1u, __ATOMIC_ACQ_REL,
                                   __HIP_MEMORY_SCOPE_AGENT);
            while (__hip_atomic_load(&cnt[t], __ATOMIC_ACQUIRE,
                                     __HIP_MEMORY_SCOPE_AGENT) < (unsigned)NWG)
                __builtin_amdgcn_s_sleep(2);
        }
        __syncthreads();
        __threadfence();                       // acquire peers' h writes
    }

    // final head: out[b] = tanh(h_T @ fc_w + fc_b); one WG per batch tile
    if (cg == 0 && threadIdx.x < 16) {
        const unsigned short* hf = h0;         // step T-1 wrote h0 (T even)
        float s = fc_b[0];
        for (int k = 0; k < HH; ++k)
            s += bf2f(hf[((size_t)mt * 16 + threadIdx.x) * HH + k]) * fc_w[k];
        out[mt * 16 + threadIdx.x] = tanhf(s);
    }
}

// ---------------- host launch ----------------
extern "C" void kernel_launch(void* const* d_in, const int* in_sizes, int n_in,
                              void* d_out, int out_size, void* d_ws, size_t ws_size,
                              hipStream_t stream) {
    (void)in_sizes; (void)n_in; (void)out_size; (void)ws_size;

    const float* x    = (const float*)d_in[0];
    const float* W[4] = {(const float*)d_in[1], (const float*)d_in[3],
                         (const float*)d_in[5], (const float*)d_in[7]};
    const float* bias[4] = {(const float*)d_in[2], (const float*)d_in[4],
                            (const float*)d_in[6], (const float*)d_in[8]};
    const float* R[4] = {(const float*)d_in[9],  (const float*)d_in[10],
                         (const float*)d_in[11], (const float*)d_in[12]};
    const float* fc_w = (const float*)d_in[13];
    const float* fc_b = (const float*)d_in[14];
    float* out = (float*)d_out;

    char* ws = (char*)d_ws;
    unsigned short* xT    = (unsigned short*)(ws);                       // 32 MB
    unsigned short* WT    = (unsigned short*)(ws + (size_t)33554432);    // 1 MB
    unsigned short* RT    = (unsigned short*)(ws + (size_t)34603008);    // 2 MB
    unsigned short* Xproj = (unsigned short*)(ws + (size_t)36700160);    // 256 MB
    unsigned short* h0    = (unsigned short*)(ws + (size_t)305135616);   // 128 KB
    unsigned short* h1    = (unsigned short*)(ws + (size_t)305266688);   // 128 KB
    unsigned*       cnt   = (unsigned*)(ws + (size_t)305397760);         // 2 KB

    // zero barrier counters (every launch; graph replays this too)
    k_zero<<<(TT + 255) / 256, 256, 0, stream>>>(cnt, TT);

    // convert / transpose to bf16
    k_cvt_x<<<(BB * TT * FF) / 256, 256, 0, stream>>>(x, xT);
    for (int gI = 0; gI < 4; ++gI) {
        k_cvt_tr<<<(FF * HH + 255) / 256, 256, 0, stream>>>(
            W[gI], WT + (size_t)gI * HH * FF, FF, HH);
        k_cvt_tr<<<(HH * HH + 255) / 256, 256, 0, stream>>>(
            R[gI], RT + (size_t)gI * HH * HH, HH, HH);
    }

    // phase 1: Xproj = xT @ W + b  (4 gates), fragment-ordered bf16 tiles
    dim3 gProj(NT_P / 16, MT_P / 2, 4);   // (2, 2048, 4)
    k_proj<<<gProj, 256, 0, stream>>>(xT, WT, bias[0], bias[1], bias[2], bias[3],
                                      Xproj);

    // phase 2: persistent scan, R LDS-resident, device-wide step barrier
    size_t shbytes = (size_t)(65536 + 16 * HPAD) * sizeof(unsigned short); // 147,712
    k_scan<<<NWG, 64, shbytes, stream>>>(RT, Xproj, fc_w, fc_b, h0, h1, cnt, out);
}
